// GraphUnet_395136991407
// MI455X (gfx1250) — matrix operand order repaired and checked by reading, not compiled, not run on hardware
//
#include <hip/hip_runtime.h>

// GraphUnet for MI455X (gfx1250): bf16 WMMA for both the A@X SpMM and the
// feature transform; ASYNC global->LDS staging (ASYNCcnt) for the SpMM tiles;
// radix-select pooling. HBM-bound by design (~0.5 GB of A traffic at
// 23.3 TB/s ~ 22us; 28 GFLOP of bf16 WMMA is far below that).

typedef __attribute__((ext_vector_type(16))) __bf16 v16bf;
typedef __attribute__((ext_vector_type(8)))  __bf16 v8bf;
typedef __attribute__((ext_vector_type(8)))  float  v8f;

#define BN   32
#define NN   1000
#define NP   1024      // padded node count (K and M padding for WMMA tiles)
#define FD   128
#define OUTD 64

// ---------------------------------------------------------------- init / convert

__global__ void k_init_mask(const float* __restrict__ mask, const int* __restrict__ nn,
                            float* __restrict__ Mv, int* __restrict__ ncur) {
    int idx = blockIdx.x * 256 + threadIdx.x;
    if (idx >= BN * NP) return;
    int b = idx / NP, i = idx % NP;
    Mv[idx] = (i < NN) ? mask[b * NN + i] : 0.0f;
    if (i == 0) ncur[b] = nn[b];
}

__global__ void k_conv_w(const float* __restrict__ W0, const float* __restrict__ W1,
                         const float* __restrict__ W2, __bf16* __restrict__ Wb) {
    int idx = blockIdx.x * 256 + threadIdx.x;
    if (idx >= 3 * FD * FD) return;
    const float* s = (idx < FD * FD) ? W0 : (idx < 2 * FD * FD ? W1 : W2);
    Wb[idx] = (__bf16)s[idx % (FD * FD)];
}

// Convert A to bf16 (zero-padded to [NP][NP]) and compute layer-0 row sums.
__global__ __launch_bounds__(256) void k_conv_A(const float* __restrict__ A,
                                                __bf16* __restrict__ Abf,
                                                float* __restrict__ RS) {
    const int i = blockIdx.x;        // padded row 0..NP-1
    const int b = blockIdx.y;
    const int tid = threadIdx.x;
    __shared__ float red[256];
    float s = 0.0f;
    __bf16* dst = Abf + ((size_t)b * NP + i) * NP;
    if (i < NN) {
        const float* src = A + ((size_t)b * NN + i) * NN;
        for (int j = tid; j < NP; j += 256) {
            float v = (j < NN) ? src[j] : 0.0f;
            dst[j] = (__bf16)v;
            s += v;
        }
    } else {
        for (int j = tid; j < NP; j += 256) dst[j] = (__bf16)0.0f;
    }
    red[tid] = s;
    __syncthreads();
    for (int off = 128; off > 0; off >>= 1) {
        if (tid < off) red[tid] += red[tid + off];
        __syncthreads();
    }
    if (tid == 0) RS[b * NP + i] = red[0];
}

// Masked row sums for layers 1/2: RS_i = m_i * sum_j A_ij * m_j
__global__ __launch_bounds__(256) void k_rowsum(const __bf16* __restrict__ Abf,
                                                const float* __restrict__ Mv,
                                                float* __restrict__ RS) {
    const int i = blockIdx.x, b = blockIdx.y, tid = threadIdx.x;
    __shared__ float red[256];
    const __bf16* row = Abf + ((size_t)b * NP + i) * NP;
    const float* m = Mv + b * NP;
    const int j0 = tid * 4;
    float s = 0.0f;
#pragma unroll
    for (int u = 0; u < 4; ++u) s += (float)row[j0 + u] * m[j0 + u];
    red[tid] = s;
    __syncthreads();
    for (int off = 128; off > 0; off >>= 1) {
        if (tid < off) red[tid] += red[tid + off];
        __syncthreads();
    }
    if (tid == 0) RS[b * NP + i] = m[i] * red[0];
}

__global__ void k_D(const float* __restrict__ RS, float* __restrict__ Dv) {
    int idx = blockIdx.x * 256 + threadIdx.x;
    if (idx < BN * NP) Dv[idx] = rsqrtf(RS[idx] + 1.0f + 1e-5f);  // deg = rowsum + I
}

// Xt[b][c][j] = bf16(D_j * x[b][j][c])  -- K-major (transposed) so the SpMM
// B-fragment (16 contiguous K per lane) stages with vectorized/async loads.
__global__ void k_prepx(const float* __restrict__ xsrc, int srcN,
                        const float* __restrict__ Dv, __bf16* __restrict__ Xt) {
    int idx = blockIdx.x * 256 + threadIdx.x;   // = b*FD*NP + c*NP + j
    if (idx >= BN * FD * NP) return;
    int j = idx & (NP - 1);
    int c = (idx >> 10) & (FD - 1);
    int b = idx >> 17;
    float v = 0.0f;
    if (j < srcN) v = Dv[b * NP + j] * xsrc[((size_t)b * srcN + j) * FD + c];
    Xt[idx] = (__bf16)v;
}

// ---------------------------------------------------------------- WMMA SpMM
// T[b][i][c] = m_i * D_i * (A @ X)[i][c] + D_i * X[i][c]   (identity folded)
// Tile: block = 64 rows x 128 cols, 8 waves, each wave 16x64 (4 wmma tiles).
// A-tile (64x32) and X-tile (128 cols x 32 K) staged with
// GLOBAL_LOAD_ASYNC_TO_LDS_B128 (ASYNCcnt), then s_wait_asynccnt + barrier.

__global__ __launch_bounds__(256) void k_spmm(const __bf16* __restrict__ Abf,
                                              const __bf16* __restrict__ Xt,
                                              const float* __restrict__ Dv,
                                              const float* __restrict__ Mv,
                                              float* __restrict__ T) {
    const int b = blockIdx.y;
    const int rowBase = blockIdx.x * 64;
    const int tid = threadIdx.x;
    const int wave = tid >> 5, lane = tid & 31;
    const int rg = wave >> 1;            // 0..3 row sub-tile
    const int cbase = (wave & 1) * 64;   // column half

    __shared__ __align__(32) __bf16 Al[64 * 32];   // row-major [64][32]
    __shared__ __align__(32) __bf16 Xl[FD * 32];   // [c][k], K contiguous

    const __bf16* Ab = Abf + ((size_t)b * NP + rowBase) * NP;
    const __bf16* Xg = Xt + (size_t)b * FD * NP;

    v8f acc[4] = {};

    // staging assignments: A 16B/thread, X 32B/thread (two b128, offset:16)
    const int ar = tid >> 2, ac = (tid & 3) * 8;
    const int xc = tid >> 1, xk = (tid & 1) * 16;
    const unsigned ldsA = (unsigned)(size_t)&Al[ar * 32 + ac];
    const unsigned ldsX = (unsigned)(size_t)&Xl[xc * 32 + xk];
    unsigned long long gA = (unsigned long long)(size_t)&Ab[(size_t)ar * NP + ac];
    unsigned long long gX = (unsigned long long)(size_t)&Xg[(size_t)xc * NP + xk];

    for (int k0 = 0; k0 < NP; k0 += 32) {
        // async DMA of both tiles into LDS (bypasses VGPRs, tracked by ASYNCcnt)
        asm volatile("global_load_async_to_lds_b128 %0, %1, off"
                     :: "v"(ldsA), "v"(gA) : "memory");
        asm volatile("global_load_async_to_lds_b128 %0, %1, off"
                     :: "v"(ldsX), "v"(gX) : "memory");
        asm volatile("global_load_async_to_lds_b128 %0, %1, off offset:16"
                     :: "v"(ldsX), "v"(gX) : "memory");
        asm volatile("s_wait_asynccnt 0x0" ::: "memory");
        __syncthreads();

        if (k0 + 32 < NP)   // prefetch next A tile (global_prefetch_b8)
            __builtin_prefetch((const void*)(size_t)(gA + 64), 0, 1);

        // A fragment: lane<16 -> M=lane, K {0..15}; lane>=16 -> M=lane-16, K {16..31}
        v16bf afrag;
        {
            const int m = lane & 15, kb = (lane >> 4) * 16;
            const __bf16* rp = &Al[(rg * 16 + m) * 32];
            v8bf lo = *(const v8bf*)&rp[kb];
            v8bf hi = *(const v8bf*)&rp[kb + 8];
#pragma unroll
            for (int u = 0; u < 8; ++u) { afrag[u] = lo[u]; afrag[8 + u] = hi[u]; }
        }
        const int nn_ = lane & 15, kb2 = (lane >> 4) * 16;
#pragma unroll
        for (int ct = 0; ct < 4; ++ct) {
            // B fragment: lane holds column N, 16 contiguous K values
            v16bf bfrag = *(const v16bf*)&Xl[(cbase + ct * 16 + nn_) * 32 + kb2];
            acc[ct] = __builtin_amdgcn_wmma_f32_16x16x32_bf16(
                false, afrag, false, bfrag, (short)0, acc[ct], false, false);
        }
        __syncthreads();
        gA += 64;   // advance 32 halfs along K
        gX += 64;
    }

    // Epilogue: C/D layout  M = r + 8*(lane>>4), N = lane&15
#pragma unroll
    for (int ct = 0; ct < 4; ++ct) {
        const int n = cbase + ct * 16 + (lane & 15);
#pragma unroll
        for (int r = 0; r < 8; ++r) {
            const int i = rowBase + rg * 16 + r + 8 * (lane >> 4);
            const float d = Dv[b * NP + i];
            const float mm = Mv[b * NP + i];
            const float self = d * (float)Xg[(size_t)n * NP + i];   // D_i^2 * x_i
            T[((size_t)b * NP + i) * FD + n] = mm * d * acc[ct][r] + self;
        }
    }
}

// ---------------------------------------------------------------- WMMA linear
// H = relu(T @ W^T + b) * mask ; W held LDS-resident (32KB bf16).

__global__ __launch_bounds__(256) void k_linear(const float* __restrict__ T,
                                                const __bf16* __restrict__ Wb,
                                                const float* __restrict__ bias,
                                                const float* __restrict__ Mv,
                                                float* __restrict__ H) {
    const int b = blockIdx.y;
    const int iBase = blockIdx.x * 16;   // grid.x = 64 -> all 1024 padded rows written
    const int tid = threadIdx.x;
    const int wave = tid >> 5, lane = tid & 31;

    __shared__ __align__(32) __bf16 Wl[FD * FD];  // [f][c]
    __shared__ __align__(32) __bf16 Tl[16 * 32];

    {   // stage full weight matrix: 64 halfs/thread
        const int base = tid * 64;
#pragma unroll
        for (int u = 0; u < 8; ++u)
            *(v8bf*)&Wl[base + u * 8] = *(const v8bf*)&Wb[base + u * 8];
    }
    __syncthreads();

    v8f acc = {};
    const int tr = tid >> 4, tc = (tid & 15) * 2;
    for (int kk = 0; kk < FD; kk += 32) {
        const size_t trow = ((size_t)b * NP + iBase + tr) * FD + kk;
        Tl[tr * 32 + tc]     = (__bf16)T[trow + tc];
        Tl[tr * 32 + tc + 1] = (__bf16)T[trow + tc + 1];
        __syncthreads();
        v16bf afrag;
        {
            const int m = lane & 15, kb = (lane >> 4) * 16;
            const __bf16* rp = &Tl[m * 32];
            v8bf lo = *(const v8bf*)&rp[kb];
            v8bf hi = *(const v8bf*)&rp[kb + 8];
#pragma unroll
            for (int u = 0; u < 8; ++u) { afrag[u] = lo[u]; afrag[8 + u] = hi[u]; }
        }
        const int f = wave * 16 + (lane & 15);
        v16bf bfrag = *(const v16bf*)&Wl[f * FD + kk + (lane >> 4) * 16];
        acc = __builtin_amdgcn_wmma_f32_16x16x32_bf16(
            false, afrag, false, bfrag, (short)0, acc, false, false);
        __syncthreads();
    }
    const int f = wave * 16 + (lane & 15);
    const float bf_ = bias[f];
#pragma unroll
    for (int r = 0; r < 8; ++r) {
        const int i = iBase + r + 8 * (lane >> 4);
        float h = acc[r] + bf_;
        h = h > 0.0f ? h : 0.0f;
        H[((size_t)b * NP + i) * FD + f] = h * Mv[b * NP + i];
    }
}

// ---------------------------------------------------------------- pooling

__global__ __launch_bounds__(256) void k_score(const float* __restrict__ H,
                                               const float* __restrict__ p,
                                               float* __restrict__ Y) {
    const int b = blockIdx.x, tid = threadIdx.x;
    float sq = 0.0f;
    for (int f = 0; f < FD; ++f) sq += p[f] * p[f];
    const float inv = rsqrtf(sq);
    for (int i = tid; i < NP; i += 256) {
        float d = 0.0f;
        if (i < NN) {
            const float* row = H + ((size_t)b * NP + i) * FD;
            for (int f = 0; f < FD; ++f) d += row[f] * p[f];
        }
        Y[b * NP + i] = d * inv;
    }
}

// Per-graph radix-select of the k-th smallest valid score; drop the k lowest
// (ties broken by node index via shared-memory scan). Updates mask in place.
__global__ __launch_bounds__(1024) void k_select(const float* __restrict__ Y,
                                                 float* __restrict__ Mv,
                                                 int* __restrict__ ncur,
                                                 float keep_ratio,
                                                 float* __restrict__ G) {
    const int b = blockIdx.x, t = threadIdx.x;
    __shared__ int cnt;
    __shared__ int sc[1024];
    const float yv = Y[b * NP + t];
    const float mv = Mv[b * NP + t];
    const bool valid = (mv > 0.0f) && (t < NN);
    unsigned u = __float_as_uint(yv);
    const unsigned key = (u & 0x80000000u) ? ~u : (u | 0x80000000u);  // order-preserving
    const int k = (int)((float)ncur[b] * (1.0f - keep_ratio));        // N_remove (uniform)
    bool removed = false;
    int kr = k;
    if (k > 0) {
        unsigned pfx = 0, dmask = 0;
        for (int bit = 31; bit >= 0; --bit) {
            const unsigned bm = 1u << bit;
            const bool cond = valid && ((key & dmask) == pfx) && ((key & bm) == 0u);
            if (t == 0) cnt = 0;
            __syncthreads();
            if (cond) atomicAdd(&cnt, 1);
            __syncthreads();
            const int c0 = cnt;
            __syncthreads();
            if (kr >= c0) { pfx |= bm; kr -= c0; }
            dmask |= bm;
        }
        const unsigned Tk = pfx;   // k-th smallest key; kr = k - #{key < Tk}
        const int flag = (valid && key == Tk) ? 1 : 0;
        sc[t] = flag;
        __syncthreads();
        for (int off = 1; off < 1024; off <<= 1) {
            const int v = (t >= off) ? sc[t - off] : 0;
            __syncthreads();
            sc[t] += v;
            __syncthreads();
        }
        const int eq_rank = sc[t] - flag;
        removed = valid && ((key < Tk) || (key == Tk && eq_rank < kr));
    }
    const float nm = (valid && !removed) ? 1.0f : 0.0f;
    Mv[b * NP + t] = nm;
    G[b * NP + t] = tanhf(yv) * nm;
    if (t == 0) ncur[b] -= k;
}

__global__ void k_gate(float* __restrict__ H, const float* __restrict__ G) {
    int idx = blockIdx.x * 256 + threadIdx.x;
    if (idx >= BN * NP * FD) return;
    H[idx] *= G[idx >> 7];
}

// ---------------------------------------------------------------- readout

__global__ __launch_bounds__(FD) void k_maxpool(const float* __restrict__ H,
                                                float* __restrict__ Gp) {
    const int b = blockIdx.x, f = threadIdx.x;
    float m = -3.4e38f;
    for (int i = 0; i < NN; ++i) {
        float v = H[((size_t)b * NP + i) * FD + f];
        m = v > m ? v : m;
    }
    Gp[b * FD + f] = m;
}

__global__ __launch_bounds__(OUTD) void k_fc(const float* __restrict__ Gp,
                                             const float* __restrict__ W,
                                             const float* __restrict__ bias,
                                             float* __restrict__ out) {
    const int b = blockIdx.x, o = threadIdx.x;
    float s = bias[o];
    for (int f = 0; f < FD; ++f) s += Gp[b * FD + f] * W[o * FD + f];
    out[b * OUTD + o] = s;
}

// ---------------------------------------------------------------- launch

extern "C" void kernel_launch(void* const* d_in, const int* in_sizes, int n_in,
                              void* d_out, int out_size, void* d_ws, size_t ws_size,
                              hipStream_t stream) {
    const float* x    = (const float*)d_in[0];
    const float* A    = (const float*)d_in[1];
    const float* mask = (const float*)d_in[2];
    const int*   nn   = (const int*)d_in[3];
    const float* W[3]    = {(const float*)d_in[4], (const float*)d_in[6], (const float*)d_in[8]};
    const float* bias[3] = {(const float*)d_in[5], (const float*)d_in[7], (const float*)d_in[9]};
    const float* p[2]    = {(const float*)d_in[10], (const float*)d_in[11]};
    const float* Wfc  = (const float*)d_in[12];
    const float* bfc  = (const float*)d_in[13];
    float* out = (float*)d_out;

    // workspace carve-up (~110 MB)
    char* w = (char*)d_ws;
    auto carve = [&](size_t bytes) { void* q = (void*)w; w += (bytes + 255) & ~(size_t)255; return q; };
    __bf16* Abf  = (__bf16*)carve((size_t)BN * NP * NP * 2);
    __bf16* Xt   = (__bf16*)carve((size_t)BN * FD * NP * 2);
    float*  Tt   = (float*) carve((size_t)BN * NP * FD * 4);
    float*  H    = (float*) carve((size_t)BN * NP * FD * 4);
    float*  Dv   = (float*) carve((size_t)BN * NP * 4);
    float*  Mv   = (float*) carve((size_t)BN * NP * 4);
    float*  Yv   = (float*) carve((size_t)BN * NP * 4);
    float*  Gv   = (float*) carve((size_t)BN * NP * 4);
    float*  RS   = (float*) carve((size_t)BN * NP * 4);
    float*  Gp   = (float*) carve((size_t)BN * FD * 4);
    __bf16* Wb   = (__bf16*)carve((size_t)3 * FD * FD * 2);
    int*    ncur = (int*)   carve((size_t)BN * 4);

    k_init_mask<<<(BN * NP + 255) / 256, 256, 0, stream>>>(mask, nn, Mv, ncur);
    k_conv_w<<<(3 * FD * FD + 255) / 256, 256, 0, stream>>>(W[0], W[1], W[2], Wb);
    k_conv_A<<<dim3(NP, BN), 256, 0, stream>>>(A, Abf, RS);

    for (int l = 0; l < 3; ++l) {
        if (l > 0) k_rowsum<<<dim3(NP, BN), 256, 0, stream>>>(Abf, Mv, RS);
        k_D<<<(BN * NP + 255) / 256, 256, 0, stream>>>(RS, Dv);
        if (l == 0) k_prepx<<<(BN * FD * NP) / 256, 256, 0, stream>>>(x, NN, Dv, Xt);
        else        k_prepx<<<(BN * FD * NP) / 256, 256, 0, stream>>>(H, NP, Dv, Xt);
        k_spmm<<<dim3(NP / 64, BN), 256, 0, stream>>>(Abf, Xt, Dv, Mv, Tt);
        k_linear<<<dim3(NP / 16, BN), 256, 0, stream>>>(Tt, Wb + (size_t)l * FD * FD, bias[l], Mv, H);
        if (l < 2) {
            k_score<<<BN, 256, 0, stream>>>(H, p[l], Yv);
            k_select<<<BN, 1024, 0, stream>>>(Yv, Mv, ncur, 0.8f, Gv);
            k_gate<<<(BN * NP * FD) / 256, 256, 0, stream>>>(H, Gv);
        }
    }
    k_maxpool<<<BN, FD, 0, stream>>>(H, Gp);
    k_fc<<<BN, OUTD, 0, stream>>>(Gp, Wfc, bfc, out);
}
